// vanilla_recurrent_39376260170351
// MI455X (gfx1250) — matrix operand order repaired and checked by reading, not compiled, not run on hardware
//
#include <hip/hip_runtime.h>
#include <hip/hip_bf16.h>

typedef __bf16 v8bf  __attribute__((ext_vector_type(8)));
typedef __bf16 v16bf __attribute__((ext_vector_type(16)));
typedef float  v8f   __attribute__((ext_vector_type(8)));

#define N_INP  512
#define N_HID  1024
#define BATCH  64
#define TSTEPS 512

#define WMMA_BF16(a, b, c) \
  __builtin_amdgcn_wmma_f32_16x16x32_bf16(false, (a), false, (b), (short)0, (c), false, false)

// ---------------- prep kernels ----------------
__global__ void cvt_f32_bf16(const float* __restrict__ src, __bf16* __restrict__ dst, int n) {
  int i = blockIdx.x * blockDim.x + threadIdx.x;
  if (i < n) dst[i] = (__bf16)src[i];
}

// src: rows x cols (row-major f32) -> dst: cols x rows (row-major bf16)
__global__ void transpose_f32_bf16(const float* __restrict__ src, __bf16* __restrict__ dst,
                                   int rows, int cols) {
  int i = blockIdx.x * blockDim.x + threadIdx.x;
  if (i >= rows * cols) return;
  int r = i / cols, c = i - r * cols;
  dst[c * rows + r] = (__bf16)src[i];
}

// A-fragment (16x32 bf16): lane holds 16 contiguous K at base
__device__ __forceinline__ v16bf load_a_frag(const __bf16* base) {
  const v8bf* p = reinterpret_cast<const v8bf*>(base);
  v8bf lo = p[0];   // K .. K+7
  v8bf hi = p[1];   // K+8 .. K+15
  return __builtin_shufflevector(lo, hi, 0,1,2,3,4,5,6,7,8,9,10,11,12,13,14,15);
}

// ---------------- phase 1: xh = inputs @ W_xh + b  (writes fp32 into d_out) ----------------
// 32x32 output tile per wave: 2 A-frags x 2 B-frags -> 4 WMMAs per K-step, 2 loads/WMMA.
// A: (32768 x 512) bf16 row-major; WT: (1024 x 512) bf16 = W_xh^T; out: (32768 x 1024) f32
__global__ void gemm_xh(const __bf16* __restrict__ A,
                        const __bf16* __restrict__ WT,
                        const float*  __restrict__ bias,
                        float* __restrict__ out)
{
  const int lane = threadIdx.x & 31;
  const int sel  = lane >> 4;          // 0 or 1
  const int lr   = lane & 15;
  const int w    = blockIdx.x * (blockDim.x >> 5) + (threadIdx.x >> 5);
  const int nT   = w & 31;             // 32 tiles of 32 columns
  const int mT   = w >> 5;             // 1024 tiles of 32 rows

  const int r0 = mT * 32 + lr;
  const int n0 = nT * 32 + lr;

  v8f c00 = {}, c01 = {}, c10 = {}, c11 = {};
  const __bf16* a0 = A  + (long)r0 * N_INP + sel * 16;
  const __bf16* a1 = a0 + 16 * N_INP;
  const __bf16* b0 = WT + (long)n0 * N_INP + sel * 16;
  const __bf16* b1 = b0 + 16 * N_INP;
  #pragma unroll 2
  for (int kk = 0; kk < N_INP; kk += 32) {
    v16bf af0 = load_a_frag(a0 + kk);
    v16bf af1 = load_a_frag(a1 + kk);
    v16bf bf0 = *reinterpret_cast<const v16bf*>(b0 + kk);
    v16bf bf1 = *reinterpret_cast<const v16bf*>(b1 + kk);
    c00 = WMMA_BF16(af0, bf0, c00);
    c01 = WMMA_BF16(af0, bf1, c01);
    c10 = WMMA_BF16(af1, bf0, c10);
    c11 = WMMA_BF16(af1, bf1, c11);
  }
  const float bv0 = bias[n0];
  const float bv1 = bias[n0 + 16];
  float* o0 = out + (long)(mT * 32 + sel * 8) * N_HID + n0;       // M rows 0..15 of tile
  float* o1 = out + (long)(mT * 32 + 16 + sel * 8) * N_HID + n0;  // M rows 16..31 of tile
  #pragma unroll
  for (int v = 0; v < 8; ++v) {
    o0[(long)v * N_HID]      = c00[v] + bv0;
    o0[(long)v * N_HID + 16] = c01[v] + bv1;
    o1[(long)v * N_HID]      = c10[v] + bv0;
    o1[(long)v * N_HID + 16] = c11[v] + bv1;
  }
}

// ---------------- phase 2: H_new = tanh(xh_t + H @ W_hh), fused in-place on d_out ----------------
// Hin: (64 x 1024) bf16; WT: (1024 x 1024) bf16 = W_hh^T; xh: d_out base (B,T,H) f32
__global__ void rnn_step(const __bf16* __restrict__ Hin,
                         const __bf16* __restrict__ WT,
                         float* __restrict__ xh,
                         __bf16* __restrict__ Hout,
                         float* __restrict__ hfinal,   // non-null only on last step
                         int t)
{
  const int lane = threadIdx.x & 31;
  const int sel  = lane >> 4;
  const int lr   = lane & 15;
  const int w    = blockIdx.x * (blockDim.x >> 5) + (threadIdx.x >> 5);  // 0..255
  const int nT   = w & 63;
  const int mT   = w >> 6;             // 0..3

  const int row = mT * 16 + lr;
  const int n   = nT * 16 + lr;

  v8f c = {};
  const __bf16* aRow = Hin + (long)row * N_HID + sel * 16;
  const __bf16* bRow = WT  + (long)n   * N_HID + sel * 16;
  #pragma unroll 4
  for (int kk = 0; kk < N_HID; kk += 32) {
    v16bf a = load_a_frag(aRow + kk);
    v16bf b = *reinterpret_cast<const v16bf*>(bRow + kk);
    c = WMMA_BF16(a, b, c);
  }
  #pragma unroll
  for (int v = 0; v < 8; ++v) {
    const int  bi  = mT * 16 + sel * 8 + v;                      // batch index
    const long idx = ((long)bi * TSTEPS + t) * N_HID + n;
    const float val = tanhf(xh[idx] + c[v]);
    xh[idx] = val;
    Hout[bi * N_HID + n] = (__bf16)val;
    if (hfinal) hfinal[bi * N_HID + n] = val;
  }
}

// ---------------- launcher ----------------
extern "C" void kernel_launch(void* const* d_in, const int* in_sizes, int n_in,
                              void* d_out, int out_size, void* d_ws, size_t ws_size,
                              hipStream_t stream) {
  const float* inputs = (const float*)d_in[0];   // (64, 512, 512)
  const float* H0     = (const float*)d_in[1];   // (64, 1024)
  const float* W_xh   = (const float*)d_in[2];   // (512, 1024)
  const float* W_hh   = (const float*)d_in[3];   // (1024, 1024)
  const float* b_h    = (const float*)d_in[4];   // (1024,)
  float* out = (float*)d_out;                    // outputs (B,T,H) ++ H_final (B,H)

  char* ws = (char*)d_ws;
  __bf16* Abf   = (__bf16*)(ws);                 // 32768*512*2  = 33554432 B
  __bf16* WTxh  = (__bf16*)(ws + 33554432);      // 1024*512*2   =  1048576 B
  __bf16* WThh  = (__bf16*)(ws + 34603008);      // 1024*1024*2  =  2097152 B
  __bf16* Hbuf0 = (__bf16*)(ws + 36700160);      // 64*1024*2    =   131072 B
  __bf16* Hbuf1 = (__bf16*)(ws + 36831232);      // 64*1024*2    =   131072 B

  // --- prep: bf16 conversion + weight transposes ---
  const int nA = BATCH * TSTEPS * N_INP;
  cvt_f32_bf16<<<(nA + 255) / 256, 256, 0, stream>>>(inputs, Abf, nA);
  transpose_f32_bf16<<<(N_INP * N_HID + 255) / 256, 256, 0, stream>>>(W_xh, WTxh, N_INP, N_HID);
  transpose_f32_bf16<<<(N_HID * N_HID + 255) / 256, 256, 0, stream>>>(W_hh, WThh, N_HID, N_HID);
  const int nH = BATCH * N_HID;
  cvt_f32_bf16<<<(nH + 255) / 256, 256, 0, stream>>>(H0, Hbuf0, nH);

  // --- phase 1: 1024x32 = 32768 waves, one 32x32 tile per wave, 4 waves/block ---
  gemm_xh<<<(1024 * 32) / 4, 128, 0, stream>>>(Abf, WTxh, b_h, out);

  // --- phase 2: 512 sequential steps, 256 tiles/step, ping-pong H ---
  float* hfin = out + (long)BATCH * TSTEPS * N_HID;
  __bf16* hb[2] = { Hbuf0, Hbuf1 };
  for (int t = 0; t < TSTEPS; ++t) {
    rnn_step<<<64, 128, 0, stream>>>(hb[t & 1], WThh, out, hb[(t + 1) & 1],
                                     (t == TSTEPS - 1) ? hfin : nullptr, t);
  }
}
